// LuminanceAwareMHSA_75024488727269
// MI455X (gfx1250) — matrix operand order repaired and checked by reading, not compiled, not run on hardware
//
#include <hip/hip_runtime.h>

typedef __attribute__((ext_vector_type(16))) _Float16 v16h;
typedef __attribute__((ext_vector_type(8)))  _Float16 h8;
typedef __attribute__((ext_vector_type(8)))  float    v8f;
typedef int v4i_ __attribute__((vector_size(16)));

static constexpr int HEADS  = 8;
static constexpr int DH     = 64;
static constexpr int DIM    = 256;
static constexpr int INNER  = 512;
static constexpr int HIDDEN = 256;
static constexpr int HH     = 64;
static constexpr int WW     = 64;
static constexpr int NPIX   = HH * WW; // 4096

#if __has_builtin(__builtin_amdgcn_global_load_async_to_lds_b128) && \
    __has_builtin(__builtin_amdgcn_s_wait_asynccnt)
#define HAS_ASYNC_LDS 1
#define GPTR(p) ((__attribute__((address_space(1))) v4i_*)(p))
#define LPTR(p) ((__attribute__((address_space(3))) v4i_*)(p))
#endif

union Frag { v16h v; h8 c[2]; _Float16 h[16]; };

__device__ __forceinline__ h8 ld8(const _Float16* p) { return *(const h8*)p; }

#define WMMA_F16(A, B, C) \
  __builtin_amdgcn_wmma_f32_16x16x32_f16(false, (A), false, (B), (short)0, (C), false, false)

// ---------------------------- tiny helper kernels ----------------------------
__global__ void k_zero(float* dst, int n) {
  int i = blockIdx.x * blockDim.x + threadIdx.x;
  if (i < n) dst[i] = 0.0f;
}

__global__ void k_cvt_f16(const float* __restrict__ src, _Float16* __restrict__ dst, int n) {
  int i = blockIdx.x * blockDim.x + threadIdx.x;
  if (i < n) dst[i] = (_Float16)src[i];
}

// x [DIM][NPIX] f32 -> xT [NPIX][DIM] f16 (channel-innermost for B fragments)
__global__ void k_cvt_xT(const float* __restrict__ x, _Float16* __restrict__ xT) {
  int gid = blockIdx.x * blockDim.x + threadIdx.x;
  int c = gid & (DIM - 1);
  int n = gid >> 8;
  xT[(size_t)n * DIM + c] = (_Float16)x[(size_t)c * NPIX + n];
}

// conv1: h1[oc][p] = relu(conv3x3(luma) + b)
__global__ void k_conv1(const float* __restrict__ luma, const float* __restrict__ w,
                        const float* __restrict__ b, float* __restrict__ h1) {
  int gid = blockIdx.x * blockDim.x + threadIdx.x;
  int oc = gid >> 12;
  int p  = gid & (NPIX - 1);
  int py = p >> 6, px = p & 63;
  float acc = b[oc];
#pragma unroll
  for (int ky = 0; ky < 3; ++ky) {
    int y = py + ky - 1;
    if ((unsigned)y > 63u) continue;
#pragma unroll
    for (int kx = 0; kx < 3; ++kx) {
      int x = px + kx - 1;
      if ((unsigned)x > 63u) continue;
      acc += luma[y * WW + x] * w[oc * 9 + ky * 3 + kx];
    }
  }
  h1[gid] = fmaxf(acc, 0.0f);
}

// conv2 + relu + spatial mean -> hm[oc]
__global__ void k_conv2mean(const float* __restrict__ h1, const float* __restrict__ w,
                            const float* __restrict__ b, float* __restrict__ hm) {
  int oc = blockIdx.y;
  int p  = blockIdx.x * blockDim.x + threadIdx.x;
  int py = p >> 6, px = p & 63;
  float acc = b[oc];
  for (int ic = 0; ic < HIDDEN; ++ic) {
    const float* base  = h1 + (size_t)ic * NPIX;
    const float* wbase = w + ((size_t)oc * HIDDEN + ic) * 9;
#pragma unroll
    for (int ky = 0; ky < 3; ++ky) {
      int y = py + ky - 1;
      if ((unsigned)y > 63u) continue;
#pragma unroll
      for (int kx = 0; kx < 3; ++kx) {
        int x = px + kx - 1;
        if ((unsigned)x > 63u) continue;
        acc += base[y * WW + x] * wbase[ky * 3 + kx];
      }
    }
  }
  acc = fmaxf(acc, 0.0f);
  __shared__ float red[256];
  red[threadIdx.x] = acc;
  __syncthreads();
  for (int s = 128; s > 0; s >>= 1) {
    if (threadIdx.x < s) red[threadIdx.x] += red[threadIdx.x + s];
    __syncthreads();
  }
  if (threadIdx.x == 0) atomicAdd(&hm[oc], red[0] * (1.0f / (float)NPIX));
}

__global__ void k_film(const float* __restrict__ w, const float* __restrict__ b,
                       const float* __restrict__ hm, float* __restrict__ film) {
  int o = blockIdx.x * blockDim.x + threadIdx.x;
  if (o >= INNER) return;
  float acc = b[o];
  for (int c = 0; c < HIDDEN; ++c) acc += hm[c] * w[o * HIDDEN + c];
  film[o] = acc;
}

__global__ void k_pool(const float* __restrict__ luma, float* __restrict__ pooled,
                       float* __restrict__ psum) {
  int p = blockIdx.x * blockDim.x + threadIdx.x;
  int py = p >> 6, px = p & 63;
  float s = 0.0f;
#pragma unroll
  for (int ky = 0; ky < 3; ++ky) {
    int y = py + ky - 1;
    if ((unsigned)y > 63u) continue;
#pragma unroll
    for (int kx = 0; kx < 3; ++kx) {
      int x = px + kx - 1;
      if ((unsigned)x > 63u) continue;
      s += 1.0f - luma[y * WW + x];
    }
  }
  s *= (1.0f / 9.0f);
  pooled[p] = s;
  __shared__ float red[256];
  red[threadIdx.x] = s;
  __syncthreads();
  for (int t = 128; t > 0; t >>= 1) {
    if (threadIdx.x < t) red[threadIdx.x] += red[threadIdx.x + t];
    __syncthreads();
  }
  if (threadIdx.x == 0) atomicAdd(psum, red[0] * (1.0f / (float)NPIX));
}

__global__ void k_keybias(const float* __restrict__ pooled, const float* __restrict__ psum,
                          const float* __restrict__ alpha, float* __restrict__ biasKey) {
  int p = blockIdx.x * blockDim.x + threadIdx.x;
  biasKey[p] = alpha[0] * (pooled[p] - psum[0]);
}

// --------------------- WMMA GEMM: QKV projection + FiLM ----------------------
__global__ __launch_bounds__(256) void k_gemm_qkv(
    const _Float16* __restrict__ Wh,   // [INNER][DIM]
    const _Float16* __restrict__ xT,   // [NPIX][DIM]
    const float* __restrict__ bias,
    const float* __restrict__ fg, const float* __restrict__ fb,
    _Float16* __restrict__ Out, float scale, int vmode) {
  int wave = threadIdx.x >> 5, lane = threadIdx.x & 31;
  int strip = blockIdx.x * (blockDim.x >> 5) + wave;  // 32 m-tiles x 64 n-strips
  int m0 = (strip >> 6) * 16;
  int n0 = (strip & 63) * 64;
  int col = lane & 15, hi = lane >> 4;

  Frag a[2], b[2][4];
  auto loadAB = [&](int kc, int s) {
    const _Float16* arow = Wh + (size_t)(m0 + col) * DIM + kc;
    a[s].c[0] = ld8(arow + 16 * hi);
    a[s].c[1] = ld8(arow + 16 * hi + 8);
    const _Float16* bbase = xT + (size_t)(n0 + col) * DIM + kc + 16 * hi;
#pragma unroll
    for (int j = 0; j < 4; ++j) {
      b[s][j].c[0] = ld8(bbase + j * 16 * DIM);
      b[s][j].c[1] = ld8(bbase + j * 16 * DIM + 8);
    }
  };
  v8f c[4] = {};
  loadAB(0, 0);
  for (int kc = 0; kc < DIM; kc += 32) {
    int s = (kc >> 5) & 1;
    if (kc + 32 < DIM) loadAB(kc + 32, s ^ 1);
#pragma unroll
    for (int j = 0; j < 4; ++j) c[j] = WMMA_F16(a[s].v, b[s][j].v, c[j]);
  }
#pragma unroll
  for (int r = 0; r < 8; ++r) {
    int o = m0 + r + (hi << 3);
    float g = fg[o], bb = bias[o], f = fb[o];
    int hd = o >> 6, d = o & 63;
#pragma unroll
    for (int j = 0; j < 4; ++j) {
      int n = n0 + 16 * j + col;
      float y = (g * (c[j][r] + bb) + f) * scale;
      if (vmode)
        Out[((size_t)hd * DH + d) * NPIX + n] = (_Float16)y;
      else
        Out[((size_t)hd * NPIX + n) * DH + d] = (_Float16)y;
    }
  }
}

// ---------------- flash attention, block-cooperative K/V staging -------------
__global__ __launch_bounds__(256) void k_attn(
    const _Float16* __restrict__ Qh,   // [HEADS][NPIX][DH], pre-scaled
    const _Float16* __restrict__ Kh,   // [HEADS][NPIX][DH]
    const _Float16* __restrict__ Vt,   // [HEADS][DH][NPIX]
    const float* __restrict__ biasKey,
    _Float16* __restrict__ On) {       // [NPIX][INNER]
  __shared__ __align__(16) _Float16 kbuf[2][32 * 64];  // 32 keys x 64 dh
  __shared__ __align__(16) _Float16 vbuf[2][64 * 32];  // 64 dh x 32 keys
  __shared__ __align__(16) _Float16 ldsP[8][2][16 * 32];
  int tid = threadIdx.x;
  int wave = tid >> 5, lane = tid & 31;
  int head = blockIdx.x >> 4;                          // 16 blocks per head
  int q0 = ((blockIdx.x & 15) * 8 + wave) * 32;
  int col = lane & 15, hi = lane >> 4;

  const _Float16* Kbase = Kh + (size_t)head * NPIX * DH;
  const _Float16* Vbase = Vt + (size_t)head * DH * NPIX;
  int koff = tid * 8;                                  // K tile is contiguous 4KB
  int vrow = tid >> 2, vcol = (tid & 3) * 8;           // V tile 64 rows x 32 halves

  // Q fragments (reused across all key steps)
  Frag aq[2][2];
#pragma unroll
  for (int qt = 0; qt < 2; ++qt) {
    const _Float16* qrow = Qh + ((size_t)head * NPIX + q0 + qt * 16 + col) * DH;
    aq[qt][0].c[0] = ld8(qrow + 16 * hi);
    aq[qt][0].c[1] = ld8(qrow + 16 * hi + 8);
    aq[qt][1].c[0] = ld8(qrow + 32 + 16 * hi);
    aq[qt][1].c[1] = ld8(qrow + 40 + 16 * hi);
  }

  float m_i[2][8], l_i[2][8];
  v8f o0[2] = {{}, {}}, o1[2] = {{}, {}}, o2[2] = {{}, {}}, o3[2] = {{}, {}};
#pragma unroll
  for (int qt = 0; qt < 2; ++qt)
#pragma unroll
    for (int r = 0; r < 8; ++r) { m_i[qt][r] = -1e30f; l_i[qt][r] = 0.0f; }

  const int NSTEPS = NPIX / 32;

#ifdef HAS_ASYNC_LDS
  auto stage_async = [&](int key0, int buf) {
    __builtin_amdgcn_global_load_async_to_lds_b128(
        GPTR(Kbase + (size_t)key0 * DH + koff),
        LPTR(&kbuf[buf][koff]), 0, 0);
    __builtin_amdgcn_global_load_async_to_lds_b128(
        GPTR(Vbase + (size_t)vrow * NPIX + key0 + vcol),
        LPTR(&vbuf[buf][vrow * 32 + vcol]), 0, 0);
  };
  stage_async(0, 0);
#else
  h8 kreg = ld8(Kbase + koff);
  h8 vreg = ld8(Vbase + (size_t)vrow * NPIX + vcol);
  h8 kreg2 = kreg, vreg2 = vreg;
#endif

  for (int step = 0; step < NSTEPS; ++step) {
    int cur = step & 1;
    int key0 = step * 32;
#ifdef HAS_ASYNC_LDS
    if (step + 1 < NSTEPS) {               // prefetch next tile into other buffer
      stage_async(key0 + 32, cur ^ 1);
      __builtin_amdgcn_s_wait_asynccnt(2); // current tile's 2 chunks landed
    } else {
      __builtin_amdgcn_s_wait_asynccnt(0);
    }
    __syncthreads();                       // everyone's chunks of buf[cur] visible
#else
    if (step + 1 < NSTEPS) {               // issue next-tile loads early
      kreg2 = ld8(Kbase + (size_t)(key0 + 32) * DH + koff);
      vreg2 = ld8(Vbase + (size_t)vrow * NPIX + key0 + 32 + vcol);
    }
    *(h8*)&kbuf[cur][koff] = kreg;
    *(h8*)&vbuf[cur][vrow * 32 + vcol] = vreg;
    __syncthreads();                       // buf[cur] visible
#endif

    // ---- S = Q K^T (from LDS K tile) ----
    const _Float16* kb = kbuf[cur];
    v8f sA[2] = {{}, {}}, sB[2] = {{}, {}};
    {
      Frag bk0, bk1;
      const _Float16* kr0 = kb + (size_t)col * 64 + 16 * hi;
      bk0.c[0] = ld8(kr0);      bk0.c[1] = ld8(kr0 + 8);
      bk1.c[0] = ld8(kr0 + 32); bk1.c[1] = ld8(kr0 + 40);
#pragma unroll
      for (int qt = 0; qt < 2; ++qt) {
        sA[qt] = WMMA_F16(aq[qt][0].v, bk0.v, sA[qt]);
        sA[qt] = WMMA_F16(aq[qt][1].v, bk1.v, sA[qt]);
      }
      const _Float16* kr1 = kb + (size_t)(16 + col) * 64 + 16 * hi;
      bk0.c[0] = ld8(kr1);      bk0.c[1] = ld8(kr1 + 8);
      bk1.c[0] = ld8(kr1 + 32); bk1.c[1] = ld8(kr1 + 40);
#pragma unroll
      for (int qt = 0; qt < 2; ++qt) {
        sB[qt] = WMMA_F16(aq[qt][0].v, bk0.v, sB[qt]);
        sB[qt] = WMMA_F16(aq[qt][1].v, bk1.v, sB[qt]);
      }
    }
    float b0 = biasKey[key0 + col];
    float b1 = biasKey[key0 + 16 + col];
#pragma unroll
    for (int qt = 0; qt < 2; ++qt) {
      float corr[8];
#pragma unroll
      for (int r = 0; r < 8; ++r) {
        sA[qt][r] += b0; sB[qt][r] += b1;
        float mx = fmaxf(sA[qt][r], sB[qt][r]);
        mx = fmaxf(mx, __shfl_xor(mx, 1));
        mx = fmaxf(mx, __shfl_xor(mx, 2));
        mx = fmaxf(mx, __shfl_xor(mx, 4));
        mx = fmaxf(mx, __shfl_xor(mx, 8));
        float mn = fmaxf(m_i[qt][r], mx);
        corr[r] = __expf(m_i[qt][r] - mn);
        sA[qt][r] = __expf(sA[qt][r] - mn);
        sB[qt][r] = __expf(sB[qt][r] - mn);
        float sum = sA[qt][r] + sB[qt][r];
        sum += __shfl_xor(sum, 1);
        sum += __shfl_xor(sum, 2);
        sum += __shfl_xor(sum, 4);
        sum += __shfl_xor(sum, 8);
        l_i[qt][r] = l_i[qt][r] * corr[r] + sum;
        m_i[qt][r] = mn;
      }
#pragma unroll
      for (int r = 0; r < 8; ++r) {
        o0[qt][r] *= corr[r]; o1[qt][r] *= corr[r];
        o2[qt][r] *= corr[r]; o3[qt][r] *= corr[r];
      }
      // stage P (C-layout -> A-layout) through per-wave LDS
#pragma unroll
      for (int r = 0; r < 8; ++r) {
        int m = r + (hi << 3);
        ldsP[wave][qt][m * 32 + col]      = (_Float16)sA[qt][r];
        ldsP[wave][qt][m * 32 + 16 + col] = (_Float16)sB[qt][r];
      }
    }
    asm volatile("s_wait_dscnt 0" ::: "memory");  // per-wave RAW on ldsP
    Frag pa[2];
#pragma unroll
    for (int qt = 0; qt < 2; ++qt) {
      pa[qt].c[0] = *(const h8*)&ldsP[wave][qt][col * 32 + 16 * hi];
      pa[qt].c[1] = *(const h8*)&ldsP[wave][qt][col * 32 + 16 * hi + 8];
    }
    // ---- O += P V (from LDS V tile) ----
    const _Float16* vbp = vbuf[cur];
    Frag bv;
    const _Float16* vr = vbp + (size_t)col * 32 + 16 * hi;
    bv.c[0] = ld8(vr); bv.c[1] = ld8(vr + 8);
    o0[0] = WMMA_F16(pa[0].v, bv.v, o0[0]);
    o0[1] = WMMA_F16(pa[1].v, bv.v, o0[1]);
    vr += 16 * 32;
    bv.c[0] = ld8(vr); bv.c[1] = ld8(vr + 8);
    o1[0] = WMMA_F16(pa[0].v, bv.v, o1[0]);
    o1[1] = WMMA_F16(pa[1].v, bv.v, o1[1]);
    vr += 16 * 32;
    bv.c[0] = ld8(vr); bv.c[1] = ld8(vr + 8);
    o2[0] = WMMA_F16(pa[0].v, bv.v, o2[0]);
    o2[1] = WMMA_F16(pa[1].v, bv.v, o2[1]);
    vr += 16 * 32;
    bv.c[0] = ld8(vr); bv.c[1] = ld8(vr + 8);
    o3[0] = WMMA_F16(pa[0].v, bv.v, o3[0]);
    o3[1] = WMMA_F16(pa[1].v, bv.v, o3[1]);

    __syncthreads();                      // all waves done with buf[cur]
#ifndef HAS_ASYNC_LDS
    kreg = kreg2; vreg = vreg2;
#endif
  }

#pragma unroll
  for (int qt = 0; qt < 2; ++qt)
#pragma unroll
    for (int r = 0; r < 8; ++r) {
      float inv = 1.0f / l_i[qt][r];
      int m = q0 + qt * 16 + r + (hi << 3);
      _Float16* orow = On + (size_t)m * INNER + head * DH;
      orow[col]      = (_Float16)(o0[qt][r] * inv);
      orow[16 + col] = (_Float16)(o1[qt][r] * inv);
      orow[32 + col] = (_Float16)(o2[qt][r] * inv);
      orow[48 + col] = (_Float16)(o3[qt][r] * inv);
    }
}

// ------------------------ WMMA GEMM: output projection ------------------------
__global__ __launch_bounds__(256) void k_gemm_oproj(
    const _Float16* __restrict__ Wph,  // [DIM][INNER]
    const _Float16* __restrict__ On,   // [NPIX][INNER]
    const float* __restrict__ bp, float* __restrict__ out) {
  int wave = threadIdx.x >> 5, lane = threadIdx.x & 31;
  int strip = blockIdx.x * (blockDim.x >> 5) + wave;  // 16 m-tiles x 64 n-strips
  int m0 = (strip >> 6) * 16;
  int n0 = (strip & 63) * 64;
  int col = lane & 15, hi = lane >> 4;

  Frag a[2], b[2][4];
  auto loadAB = [&](int kc, int s) {
    const _Float16* arow = Wph + (size_t)(m0 + col) * INNER + kc;
    a[s].c[0] = ld8(arow + 16 * hi);
    a[s].c[1] = ld8(arow + 16 * hi + 8);
    const _Float16* bbase = On + (size_t)(n0 + col) * INNER + kc + 16 * hi;
#pragma unroll
    for (int j = 0; j < 4; ++j) {
      b[s][j].c[0] = ld8(bbase + j * 16 * INNER);
      b[s][j].c[1] = ld8(bbase + j * 16 * INNER + 8);
    }
  };
  v8f c[4] = {};
  loadAB(0, 0);
  for (int kc = 0; kc < INNER; kc += 32) {
    int s = (kc >> 5) & 1;
    if (kc + 32 < INNER) loadAB(kc + 32, s ^ 1);
#pragma unroll
    for (int j = 0; j < 4; ++j) c[j] = WMMA_F16(a[s].v, b[s][j].v, c[j]);
  }
#pragma unroll
  for (int r = 0; r < 8; ++r) {
    int o = m0 + r + (hi << 3);
    float bb = bp[o];
#pragma unroll
    for (int j = 0; j < 4; ++j) {
      int n = n0 + 16 * j + col;
      out[(size_t)o * NPIX + n] = c[j][r] + bb;
    }
  }
}

// --------------------------------- launcher ----------------------------------
extern "C" void kernel_launch(void* const* d_in, const int* in_sizes, int n_in,
                              void* d_out, int out_size, void* d_ws, size_t ws_size,
                              hipStream_t stream) {
  const float* x    = (const float*)d_in[0];
  const float* luma = (const float*)d_in[1];
  const float* Wq = (const float*)d_in[2];  const float* bq = (const float*)d_in[3];
  const float* Wk = (const float*)d_in[4];  const float* bk = (const float*)d_in[5];
  const float* Wv = (const float*)d_in[6];  const float* bv = (const float*)d_in[7];
  const float* Wp = (const float*)d_in[8];  const float* bp = (const float*)d_in[9];
  const float* c1w = (const float*)d_in[10]; const float* c1b = (const float*)d_in[11];
  const float* c2w = (const float*)d_in[12]; const float* c2b = (const float*)d_in[13];
  const float* gqw = (const float*)d_in[14]; const float* gqb = (const float*)d_in[15];
  const float* bqw = (const float*)d_in[16]; const float* bqb = (const float*)d_in[17];
  const float* gkw = (const float*)d_in[18]; const float* gkb = (const float*)d_in[19];
  const float* bkw = (const float*)d_in[20]; const float* bkb = (const float*)d_in[21];
  const float* gvw = (const float*)d_in[22]; const float* gvb = (const float*)d_in[23];
  const float* bvw = (const float*)d_in[24]; const float* bvb = (const float*)d_in[25];
  const float* alpha = (const float*)d_in[26];
  float* out = (float*)d_out;

  char* ws = (char*)d_ws;
  size_t off = 0;
  auto alloc = [&](size_t bytes) { size_t o = off; off += (bytes + 255) & ~(size_t)255; return o; };

  _Float16* xT  = (_Float16*)(ws + alloc((size_t)NPIX * DIM * 2));
  _Float16* wqh = (_Float16*)(ws + alloc((size_t)INNER * DIM * 2));
  _Float16* wkh = (_Float16*)(ws + alloc((size_t)INNER * DIM * 2));
  _Float16* wvh = (_Float16*)(ws + alloc((size_t)INNER * DIM * 2));
  _Float16* wph = (_Float16*)(ws + alloc((size_t)DIM * INNER * 2));
  _Float16* qh  = (_Float16*)(ws + alloc((size_t)HEADS * NPIX * DH * 2));
  _Float16* kh  = (_Float16*)(ws + alloc((size_t)HEADS * NPIX * DH * 2));
  _Float16* vt  = (_Float16*)(ws + alloc((size_t)HEADS * DH * NPIX * 2));
  _Float16* on  = (_Float16*)(ws + alloc((size_t)NPIX * INNER * 2));
  float* h1     = (float*)(ws + alloc((size_t)HIDDEN * NPIX * 4));
  float* hm     = (float*)(ws + alloc((size_t)(HIDDEN + 1) * 4));
  float* psum   = hm + HIDDEN;
  float* film   = (float*)(ws + alloc((size_t)6 * INNER * 4));
  float* pooled = (float*)(ws + alloc((size_t)NPIX * 4));
  float* biasKey= (float*)(ws + alloc((size_t)NPIX * 4));
  float* fgq = film + 0 * INNER, *fbq = film + 1 * INNER;
  float* fgk = film + 2 * INNER, *fbk = film + 3 * INNER;
  float* fgv = film + 4 * INNER, *fbv = film + 5 * INNER;

  k_zero<<<2, 256, 0, stream>>>(hm, HIDDEN + 1);

  k_cvt_xT<<<(DIM * NPIX) / 256, 256, 0, stream>>>(x, xT);
  k_cvt_f16<<<(INNER * DIM + 255) / 256, 256, 0, stream>>>(Wq, wqh, INNER * DIM);
  k_cvt_f16<<<(INNER * DIM + 255) / 256, 256, 0, stream>>>(Wk, wkh, INNER * DIM);
  k_cvt_f16<<<(INNER * DIM + 255) / 256, 256, 0, stream>>>(Wv, wvh, INNER * DIM);
  k_cvt_f16<<<(DIM * INNER + 255) / 256, 256, 0, stream>>>(Wp, wph, DIM * INNER);

  k_conv1<<<(HIDDEN * NPIX) / 256, 256, 0, stream>>>(luma, c1w, c1b, h1);
  k_conv2mean<<<dim3(NPIX / 256, HIDDEN), 256, 0, stream>>>(h1, c2w, c2b, hm);
  k_film<<<2, 256, 0, stream>>>(gqw, gqb, hm, fgq);
  k_film<<<2, 256, 0, stream>>>(bqw, bqb, hm, fbq);
  k_film<<<2, 256, 0, stream>>>(gkw, gkb, hm, fgk);
  k_film<<<2, 256, 0, stream>>>(bkw, bkb, hm, fbk);
  k_film<<<2, 256, 0, stream>>>(gvw, gvb, hm, fgv);
  k_film<<<2, 256, 0, stream>>>(bvw, bvb, hm, fbv);

  k_pool<<<NPIX / 256, 256, 0, stream>>>(luma, pooled, psum);
  k_keybias<<<NPIX / 256, 256, 0, stream>>>(pooled, psum, alpha, biasKey);

  const float scale = 0.125f;  // DH^-0.5
  int qkvBlocks = (INNER / 16) * (NPIX / 64) / 8;  // 256
  k_gemm_qkv<<<qkvBlocks, 256, 0, stream>>>(wqh, xT, bq, fgq, fbq, qh, scale, 0);
  k_gemm_qkv<<<qkvBlocks, 256, 0, stream>>>(wkh, xT, bk, fgk, fbk, kh, 1.0f, 0);
  k_gemm_qkv<<<qkvBlocks, 256, 0, stream>>>(wvh, xT, bv, fgv, fbv, vt, 1.0f, 1);

  int attnBlocks = HEADS * (NPIX / 32) / 8;  // 128
  k_attn<<<attnBlocks, 256, 0, stream>>>(qh, kh, vt, biasKey, on);

  int opBlocks = (DIM / 16) * (NPIX / 64) / 8;  // 128
  k_gemm_oproj<<<opBlocks, 256, 0, stream>>>(wph, on, bp, out);
}
